// MultiheadAttention_18854906430201
// MI455X (gfx1250) — matrix operand order repaired and checked by reading, not compiled, not run on hardware
//
#include <hip/hip_runtime.h>
#include <hip/hip_bf16.h>
#include <stdint.h>

#define L_    1024
#define B_    16
#define D_    512
#define H_    8
#define DK_   64
#define HD_   (H_*D_)      // 4096
#define TEMP_ 30.0f

typedef __attribute__((ext_vector_type(16))) __bf16 bf16x16;
typedef __attribute__((ext_vector_type(8)))  __bf16 bf16x8;
typedef __attribute__((ext_vector_type(8)))  float  f32x8;

static __device__ __forceinline__ bf16x16 cat8(bf16x8 lo, bf16x8 hi) {
  return __builtin_shufflevector(lo, hi, 0,1,2,3,4,5,6,7,8,9,10,11,12,13,14,15);
}
static __device__ __forceinline__ f32x8 wmma_bf16(bf16x16 a, bf16x16 b, f32x8 c) {
  // D = A(16x32) * B(32x16) + C, f32 accumulate
  return __builtin_amdgcn_wmma_f32_16x16x32_bf16(false, a, false, b, (short)0, c,
                                                 false, false);
}
// async DMA: 16 bytes global -> LDS per lane, tracked by ASYNCcnt
static __device__ __forceinline__ void async_copy16(const __bf16* g, __bf16* l) {
  uint32_t laddr = (uint32_t)(uintptr_t)l;          // low 32 bits = LDS byte offset
  uint64_t gaddr = (uint64_t)(uintptr_t)g;
  asm volatile("global_load_async_to_lds_b128 %0, %1, off"
               :: "v"(laddr), "v"(gaddr) : "memory");
}

// ---------------------------------------------------------------------------
// Kernel 0: streaming fp32 -> bf16 convert (query / key / Wk_w, converted ONCE
// instead of per-head inside the projection GEMM). 8 elems per thread.
// ---------------------------------------------------------------------------
__global__ __launch_bounds__(256)
void cvt_bf16_kernel(const float* __restrict__ src, __bf16* __restrict__ dst)
{
  const size_t i = ((size_t)blockIdx.x * 256 + threadIdx.x) * 8;
  f32x8 v = *(const f32x8*)(src + i);
  bf16x8 o;
  #pragma unroll
  for (int k = 0; k < 8; ++k) o[k] = (__bf16)v[k];
  *(bf16x8*)(dst + i) = o;
}

// ---------------------------------------------------------------------------
// Kernel 1: per-head projection + L2 normalize -> bf16 wq / wk in workspace.
// All GEMM operands already bf16 -> no converts in the hot loop.
// TEMP is folded into wq so the attention inner loop needs no logit scaling.
// grid: (L/128, B, 2H). One wave = 16 rows x DK(=64) output tile.
// ---------------------------------------------------------------------------
__global__ __launch_bounds__(256)
void proj_norm_kernel(const __bf16* __restrict__ Xq, const __bf16* __restrict__ Xk,
                      const __bf16* __restrict__ Wbf, const float* __restrict__ Wk_b,
                      __bf16* __restrict__ wq, __bf16* __restrict__ wk)
{
  const int lane   = threadIdx.x & 31;
  const int wave   = threadIdx.x >> 5;
  const int lane16 = lane & 15;
  const int half   = lane >> 4;

  const int b = blockIdx.y;
  const int z = blockIdx.z;
  const int h = z & (H_ - 1);
  const __bf16* X   = (z < H_) ? Xq : Xk;
  __bf16*    Wout   = (z < H_) ? wq : wk;
  const float postS = (z < H_) ? TEMP_ : 1.0f;     // fold temperature into wq

  const int l0 = (blockIdx.x * 8 + wave) * 16;
  const __bf16* xrow = X + ((size_t)(l0 + lane16) * B_ + b) * D_;  // A: lane16 = M

  f32x8 acc[4] = {};                                               // 4 N-tiles of DK=64
  #pragma unroll 4
  for (int s = 0; s < D_; s += 32) {
    // A fragment, contiguous K order (matches B): e -> K = s + half*16 + e
    bf16x16 a = cat8(*(const bf16x8*)(xrow + s + half * 16),
                     *(const bf16x8*)(xrow + s + half * 16 + 8));
    #pragma unroll
    for (int t = 0; t < 4; ++t) {
      // B fragment: lane16 = N -> W row (h*64 + t*16 + lane16); e -> K = half*16 + e
      const __bf16* wrow =
          Wbf + ((size_t)h * DK_ + t * 16 + lane16) * D_ + s + half * 16;
      acc[t] = wmma_bf16(a, *(const bf16x16*)wrow, acc[t]);
    }
  }
  // bias, row sum-of-squares (row j lives in element j, columns across the 16-lane half)
  float ss[8];
  #pragma unroll
  for (int j = 0; j < 8; ++j) ss[j] = 0.f;
  #pragma unroll
  for (int t = 0; t < 4; ++t) {
    float bias = Wk_b[h * DK_ + t * 16 + lane16];
    #pragma unroll
    for (int j = 0; j < 8; ++j) { acc[t][j] += bias; ss[j] += acc[t][j] * acc[t][j]; }
  }
  #pragma unroll
  for (int j = 0; j < 8; ++j) {
    float v = ss[j];
    v += __shfl_xor(v, 1, 32); v += __shfl_xor(v, 2, 32);
    v += __shfl_xor(v, 4, 32); v += __shfl_xor(v, 8, 32);
    ss[j] = postS / fmaxf(sqrtf(v), 1e-12f);
  }
  __bf16* orow = Wout + ((size_t)((h * B_ + b) * L_) + l0) * DK_;
  #pragma unroll
  for (int t = 0; t < 4; ++t)
    #pragma unroll
    for (int j = 0; j < 8; ++j)
      orow[(size_t)(j + half * 8) * DK_ + t * 16 + lane16] = (__bf16)(acc[t][j] * ss[j]);
}

// ---------------------------------------------------------------------------
// Kernel 2: V[m,b,d] fp32 -> Vt[b,d,m] bf16 (contiguous along m for B-fragments)
// ---------------------------------------------------------------------------
__global__ __launch_bounds__(128)
void vtrans_kernel(const float* __restrict__ value, __bf16* __restrict__ Vt)
{
  const int b  = blockIdx.z;
  const int d  = blockIdx.y * 16 + (threadIdx.x & 15);
  const int m0 = blockIdx.x * 64 + (threadIdx.x >> 4) * 8;
  bf16x8 v;
  #pragma unroll
  for (int i = 0; i < 8; ++i)
    v[i] = (__bf16)value[(size_t)(m0 + i) * B_ * D_ + (size_t)b * D_ + d];
  *(bf16x8*)(Vt + (size_t)b * D_ * L_ + (size_t)d * L_ + m0) = v;
}

// ---------------------------------------------------------------------------
// Kernel 3: fused flash attention. One wave = 16 l-rows x 64 d-cols output tile.
// wk / V tiles are double-buffer staged block-wide via async global->LDS DMA.
// grid: (L/128, D/64, H*B), block 256 (8 waves).
// ---------------------------------------------------------------------------
__global__ __launch_bounds__(256)
void attn_kernel(const __bf16* __restrict__ wq, const __bf16* __restrict__ wk,
                 const __bf16* __restrict__ Vt, float* __restrict__ out)
{
  __shared__ __align__(32) __bf16 wkS[2][32 * 64];   // 2 x 4KB: 32 m-rows x 64 k
  __shared__ __align__(32) __bf16 vtS[2][64 * 32];   // 2 x 4KB: 64 d-rows x 32 m
  __shared__ __align__(32) __bf16 ldsP[8][16 * 32];  // per-wave P transpose hop

  const int lane   = threadIdx.x & 31;
  const int wave   = threadIdx.x >> 5;
  const int lane16 = lane & 15;
  const int half   = lane >> 4;
  const int tid    = threadIdx.x;

  const int hb = blockIdx.z;
  const int h  = hb / B_;
  const int b  = hb % B_;
  const int d0 = blockIdx.y * 64;
  const int l0 = (blockIdx.x * 8 + wave) * 16;

  const __bf16* wkbase = wk + (size_t)((h * B_ + b) * L_) * DK_;
  const __bf16* vtb    = Vt + (size_t)b * D_ * L_;

  // Q A-fragments, K split 0..31 / 32..63 (wq already carries TEMP)
  const __bf16* qrow = wq + ((size_t)((h * B_ + b) * L_) + l0 + lane16) * DK_;
  bf16x16 aq0 = cat8(*(const bf16x8*)(qrow + half * 16),
                     *(const bf16x8*)(qrow + half * 16 + 8));
  bf16x16 aq1 = cat8(*(const bf16x8*)(qrow + 32 + half * 16),
                     *(const bf16x8*)(qrow + 32 + half * 16 + 8));

  f32x8 accO[4] = {};
  float mrun[8], lrun[8];
  #pragma unroll
  for (int j = 0; j < 8; ++j) { mrun[j] = -1e30f; lrun[j] = 0.f; }

  // stage one 32-step tile pair (wk 4KB + Vt 4KB); 2 async b128 per thread
  auto stage = [&](int buf, int m0) {
    async_copy16(wkbase + (size_t)(m0 + (tid >> 3)) * DK_ + (tid & 7) * 8,
                 &wkS[buf][tid * 8]);
    async_copy16(vtb + (size_t)(d0 + (tid >> 2)) * L_ + m0 + (tid & 3) * 8,
                 &vtS[buf][tid * 8]);
  };

  stage(0, 0);
  for (int m0 = 0; m0 < L_; m0 += 32) {
    const int cur = (m0 >> 5) & 1;
    if (m0 + 32 < L_) {
      stage(cur ^ 1, m0 + 32);                       // prefetch next tile pair
      asm volatile("s_wait_asynccnt 0x2" ::: "memory");
    } else {
      asm volatile("s_wait_asynccnt 0x0" ::: "memory");
    }
    __syncthreads();                                 // staged tiles visible

    // ---- S = wq_scaled . wk^T for columns m0..m0+31 (from LDS) ----
    const __bf16* kb = &wkS[cur][0];
    bf16x16 bk00 = *(const bf16x16*)(kb + (size_t)lane16 * 64 + half * 16);
    bf16x16 bk01 = *(const bf16x16*)(kb + (size_t)lane16 * 64 + 32 + half * 16);
    bf16x16 bk10 = *(const bf16x16*)(kb + (size_t)(16 + lane16) * 64 + half * 16);
    bf16x16 bk11 = *(const bf16x16*)(kb + (size_t)(16 + lane16) * 64 + 32 + half * 16);

    f32x8 s0 = {}, s1 = {};
    s0 = wmma_bf16(aq0, bk00, s0);
    s0 = wmma_bf16(aq1, bk01, s0);
    s1 = wmma_bf16(aq0, bk10, s1);
    s1 = wmma_bf16(aq1, bk11, s1);

    // ---- online softmax (row j of the half; columns across 16 lanes) ----
    #pragma unroll
    for (int j = 0; j < 8; ++j) {
      float a = s0[j], c = s1[j];
      float rmax = fmaxf(a, c);
      rmax = fmaxf(rmax, __shfl_xor(rmax, 1, 32));
      rmax = fmaxf(rmax, __shfl_xor(rmax, 2, 32));
      rmax = fmaxf(rmax, __shfl_xor(rmax, 4, 32));
      rmax = fmaxf(rmax, __shfl_xor(rmax, 8, 32));
      float mnew = fmaxf(mrun[j], rmax);
      float corr = __expf(mrun[j] - mnew);
      mrun[j] = mnew;
      float p0 = __expf(a - mnew);
      float p1 = __expf(c - mnew);
      float rs = p0 + p1;
      rs += __shfl_xor(rs, 1, 32); rs += __shfl_xor(rs, 2, 32);
      rs += __shfl_xor(rs, 4, 32); rs += __shfl_xor(rs, 8, 32);
      lrun[j] = lrun[j] * corr + rs;
      #pragma unroll
      for (int t = 0; t < 4; ++t) accO[t][j] *= corr;
      ldsP[wave][(j + half * 8) * 32 + lane16]      = (__bf16)p0;
      ldsP[wave][(j + half * 8) * 32 + 16 + lane16] = (__bf16)p1;
    }
    // wave-private LDS region; per-wave DS ops are in-order -> only a
    // compiler scheduling fence is needed, not a block barrier.
    __builtin_amdgcn_wave_barrier();
    const __bf16* pbase = &ldsP[wave][lane16 * 32];
    bf16x16 pA = cat8(*(const bf16x8*)(pbase + half * 16),
                      *(const bf16x8*)(pbase + half * 16 + 8));

    // ---- O += P * V^T for the 64-wide d slice (V from LDS) ----
    const __bf16* vb = &vtS[cur][0];
    #pragma unroll
    for (int t = 0; t < 4; ++t) {
      bf16x16 bv = *(const bf16x16*)(vb + (size_t)(t * 16 + lane16) * 32 + half * 16);
      accO[t] = wmma_bf16(pA, bv, accO[t]);
    }
    __syncthreads();                                 // all reads done before re-stage
  }

  // ---- epilogue: O /= l, write out[l, b, h*512 + d] ----
  float* orow = out + (size_t)l0 * B_ * HD_ + (size_t)b * HD_ + h * D_ + d0;
  #pragma unroll
  for (int j = 0; j < 8; ++j) {
    float inv = 1.0f / lrun[j];
    #pragma unroll
    for (int t = 0; t < 4; ++t)
      orow[(size_t)(j + half * 8) * B_ * HD_ + t * 16 + lane16] = accO[t][j] * inv;
  }
}

// ---------------------------------------------------------------------------
extern "C" void kernel_launch(void* const* d_in, const int* in_sizes, int n_in,
                              void* d_out, int out_size, void* d_ws, size_t ws_size,
                              hipStream_t stream) {
  const float* query = (const float*)d_in[0];
  const float* key   = (const float*)d_in[1];
  const float* value = (const float*)d_in[2];
  const float* Wk_w  = (const float*)d_in[3];
  const float* Wk_b  = (const float*)d_in[4];
  float* out = (float*)d_out;

  // workspace layout (bf16): wq | wk | Vt | Xq | Xk | Wb   (~85 MB)
  const size_t WQK = (size_t)H_ * B_ * L_ * DK_;     // 8.39M elems
  const size_t XN  = (size_t)L_ * B_ * D_;           // 8.39M elems
  const size_t WN  = (size_t)H_ * DK_ * D_;          // 262144 elems
  __bf16* wq  = (__bf16*)d_ws;
  __bf16* wk  = wq  + WQK;
  __bf16* Vt  = wk  + WQK;
  __bf16* Xqb = Vt  + XN;
  __bf16* Xkb = Xqb + XN;
  __bf16* Wb  = Xkb + XN;

  // one-time fp32 -> bf16 converts (streaming, HBM-bound)
  cvt_bf16_kernel<<<dim3(XN / 2048), 256, 0, stream>>>(query, Xqb);
  cvt_bf16_kernel<<<dim3(XN / 2048), 256, 0, stream>>>(key,   Xkb);
  cvt_bf16_kernel<<<dim3(WN / 2048), 256, 0, stream>>>(Wk_w,  Wb);

  dim3 gv(L_ / 64, D_ / 16, B_);
  vtrans_kernel<<<gv, 128, 0, stream>>>(value, Vt);

  dim3 gp(L_ / 128, B_, 2 * H_);
  proj_norm_kernel<<<gp, 256, 0, stream>>>(Xqb, Xkb, Wb, Wk_b, wq, wk);

  dim3 ga(L_ / 128, D_ / 64, H_ * B_);
  attn_kernel<<<ga, 256, 0, stream>>>(wq, wk, Vt, out);
}
